// ResidualBiMambaEncoder_34093450396625
// MI455X (gfx1250) — compile-verified, matrix operand order checked by reading, not run on hardware
//
#include <hip/hip_runtime.h>
#include <hip/hip_bf16.h>

typedef __attribute__((ext_vector_type(16))) _Float16 v16h;
typedef __attribute__((ext_vector_type(8)))  _Float16 v8h;
typedef __attribute__((ext_vector_type(8)))  float    v8f;
typedef __attribute__((ext_vector_type(4)))  int      v4i;
typedef v4i __attribute__((address_space(1))) v4i_g;   // global int4
typedef v4i __attribute__((address_space(3))) v4i_l;   // LDS int4

#define NBATCH   64
#define SEQL     512
#define MTOT     (NBATCH * SEQL)   // 32768 rows
#define DMODEL   64
#define DINNER   128
#define DSTATE   16
#define DTRANK   4
#define DBLPAD   48                // dt_rank + 2*d_state = 36, padded to 3 WMMA tiles
#define SCAN_T   32                // timesteps staged per LDS chunk
#define SCAN_NCH (SEQL / SCAN_T)   // 16 chunks

// --------------------------------------------------------------------------
// gfx1250 async global->LDS path (probe via __has_builtin; message -> stderr)
// --------------------------------------------------------------------------
#if defined(__has_builtin)
#if __has_builtin(__builtin_amdgcn_global_load_async_to_lds_b128) && \
    __has_builtin(__builtin_amdgcn_s_wait_asynccnt)
#define BM_ASYNC 1
#endif
#endif
#ifndef BM_ASYNC
#define BM_ASYNC 0
#endif

#if BM_ASYNC
#pragma message("BM: using global_load_async_to_lds_b128 + s_wait_asynccnt path")
#else
#pragma message("BM: async builtins unavailable -> synchronous LDS staging fallback")
#endif

__device__ inline void bm_cp128(const float* __restrict__ g, float* l) {
#if BM_ASYNC
  __builtin_amdgcn_global_load_async_to_lds_b128((v4i_g*)g, (v4i_l*)l, 0, 0);
#else
  *(float4*)l = *(const float4*)g;
#endif
}

// ---------------------------------------------------------------------------
// (0) Weight prep: fp32 -> f16 weight images, A = -exp(A_log)
// ---------------------------------------------------------------------------
__global__ void bm_prep_kernel(const float* WinF, const float* WinB,
                               const float* WxF,  const float* WxB,
                               const float* Wout,
                               const float* AlogF, const float* AlogB,
                               _Float16* wInH, _Float16* wXH, _Float16* wOutH,
                               float* Aneg) {
  int i = blockIdx.x * blockDim.x + threadIdx.x;
  if (i < 16384)        wInH[i]          = (_Float16)WinF[i];
  else if (i < 32768)   wInH[i]          = (_Float16)WinB[i - 16384];
  else if (i < 37376)   wXH[i - 32768]   = (_Float16)WxF[i - 32768];
  else if (i < 41984)   wXH[i - 32768]   = (_Float16)WxB[i - 37376];
  else if (i < 50176)   wOutH[i - 41984] = (_Float16)Wout[i - 41984];
  else if (i < 52224)   Aneg[i - 50176]  = -__expf(AlogF[i - 50176]);
  else if (i < 54272)   Aneg[i - 50176]  = -__expf(AlogB[i - 52224]);
}

// ---------------------------------------------------------------------------
// (1) Pixel-shuffle gather + LayerNorm over 64 channels -> xn (f16, [MTOT,64])
// ---------------------------------------------------------------------------
__global__ void bm_ln_kernel(const float* __restrict__ x,
                             const float* __restrict__ gamma,
                             const float* __restrict__ beta,
                             _Float16* __restrict__ xn) {
  __shared__ float s1[256];
  __shared__ float s2[256];
  int sub = threadIdx.x >> 6;       // position within block
  int c   = threadIdx.x & 63;       // channel
  int pos = blockIdx.x * 4 + sub;   // scan-order row (nb*512 + t)
  int nb = pos >> 9, t = pos & 511;
  int pz = (nb >> 4) & 3, ph = (nb >> 2) & 3, pw = nb & 3;
  int nz = (t >> 6) & 7,  nh = (t >> 3) & 7,  nw = t & 7;
  int lin = ((c * 32 + nz * 4 + pz) * 32 + nh * 4 + ph) * 32 + nw * 4 + pw;
  float v = x[lin];
  s1[threadIdx.x] = v;
  s2[threadIdx.x] = v * v;
  __syncthreads();
  #pragma unroll
  for (int off = 32; off >= 1; off >>= 1) {
    if (c < off) {
      s1[threadIdx.x] += s1[threadIdx.x + off];
      s2[threadIdx.x] += s2[threadIdx.x + off];
    }
    __syncthreads();
  }
  float mu  = s1[sub * 64] * (1.0f / 64.0f);
  float var = s2[sub * 64] * (1.0f / 64.0f) - mu * mu;
  float rs  = rsqrtf(var + 1e-5f);
  xn[pos * 64 + c] = (_Float16)((v - mu) * rs * gamma[c] + beta[c]);
}

// ---------------------------------------------------------------------------
// WMMA 16x32 f16 fragment load per ISA 16-bit layout (two b128 chunks)
// ---------------------------------------------------------------------------
__device__ inline v16h bm_frag_f16(const _Float16* p /* row base + kb + kc*32 */) {
  const v8h* vp = (const v8h*)p;
  v8h lo = vp[0];
  v8h hi = vp[2];      // +16 halves
  v16h f;
  #pragma unroll
  for (int j = 0; j < 8; ++j) { f[j] = lo[j]; f[8 + j] = hi[j]; }
  return f;
}

// ---------------------------------------------------------------------------
// (2) in_proj: xz[MTOT,256] = xn[MTOT,64] @ W_in^T   (dir=1 reverses seq gather)
// ---------------------------------------------------------------------------
__global__ void bm_inproj_kernel(const _Float16* __restrict__ xn,
                                 const _Float16* __restrict__ W,   // [256,64] f16
                                 float* __restrict__ xz, int rev) {
  int lane = threadIdx.x & 31;
  int wave = threadIdx.x >> 5;
  int tile = blockIdx.x * 8 + wave;        // 2048 * 16 tiles
  int mt = tile >> 4, nt = tile & 15;
  int mrow = mt * 16 + (lane & 15);        // scan-order row
  int nb = mrow >> 9, t = mrow & 511;
  int mnat = rev ? ((nb << 9) | (511 - t)) : mrow;
  int kb = (lane < 16) ? 0 : 8;
  const _Float16* arow = xn + mnat * 64 + kb;
  const _Float16* brow = W + (nt * 16 + (lane & 15)) * 64 + kb;
  v8f acc = {};
  #pragma unroll
  for (int kc = 0; kc < 2; ++kc) {
    v16h a = bm_frag_f16(arow + kc * 32);
    v16h b = bm_frag_f16(brow + kc * 32);
    acc = __builtin_amdgcn_wmma_f32_16x16x32_f16(false, a, false, b,
                                                 (short)0, acc, false, false);
  }
  int n  = nt * 16 + (lane & 15);
  int mb = mt * 16 + ((lane < 16) ? 0 : 8);
  #pragma unroll
  for (int r = 0; r < 8; ++r) xz[(mb + r) * 256 + n] = acc[r];
}

// ---------------------------------------------------------------------------
// (3) Depthwise causal conv (k=4) + bias + SiLU on x-half of xz
// ---------------------------------------------------------------------------
__global__ void bm_conv_kernel(const float* __restrict__ xz,
                               const float* __restrict__ cw,   // [128,1,4]
                               const float* __restrict__ cb,   // [128]
                               float* __restrict__ xcf, _Float16* __restrict__ xch) {
  int idx = blockIdx.x * blockDim.x + threadIdx.x;   // MTOT*128
  int m = idx >> 7, d = idx & 127;
  int t = m & 511;
  float acc = cb[d];
  #pragma unroll
  for (int i = 0; i < 4; ++i) {
    int tt = t - 3 + i;
    if (tt >= 0) acc += cw[d * 4 + i] * xz[(m - 3 + i) * 256 + d];
  }
  float s = acc * (1.0f / (1.0f + __expf(-acc)));    // SiLU
  xcf[m * 128 + d] = s;
  xch[m * 128 + d] = (_Float16)s;
}

// ---------------------------------------------------------------------------
// (4) x_proj: dbl[MTOT,48] = xconv[MTOT,128] @ W_x^T (36 valid cols, padded)
// ---------------------------------------------------------------------------
__global__ void bm_xproj_kernel(const _Float16* __restrict__ xch,
                                const _Float16* __restrict__ Wx,  // [36,128] f16
                                float* __restrict__ dbl) {
  int lane = threadIdx.x & 31;
  int wave = threadIdx.x >> 5;
  int tile = blockIdx.x * 8 + wave;        // 2048 * 3 tiles
  if (tile >= 2048 * 3) return;
  int mt = tile / 3, nt = tile % 3;
  int kb = (lane < 16) ? 0 : 8;
  int wrow = nt * 16 + (lane & 15);
  bool bval = (wrow < 36);
  const _Float16* arow = xch + (mt * 16 + (lane & 15)) * 128 + kb;
  const _Float16* brow = Wx + wrow * 128 + kb;
  v8f acc = {};
  #pragma unroll
  for (int kc = 0; kc < 4; ++kc) {
    v16h a = bm_frag_f16(arow + kc * 32);
    v16h b;
    if (bval) {
      b = bm_frag_f16(brow + kc * 32);
    } else {
      #pragma unroll
      for (int j = 0; j < 16; ++j) b[j] = (_Float16)0.0f;
    }
    acc = __builtin_amdgcn_wmma_f32_16x16x32_f16(false, a, false, b,
                                                 (short)0, acc, false, false);
  }
  int n  = nt * 16 + (lane & 15);
  int mb = mt * 16 + ((lane < 16) ? 0 : 8);
  #pragma unroll
  for (int r = 0; r < 8; ++r) dbl[(mb + r) * DBLPAD + n] = acc[r];
}

// ---------------------------------------------------------------------------
// (5) dt_proj (K=4) + softplus
// ---------------------------------------------------------------------------
__global__ void bm_dt_kernel(const float* __restrict__ dbl,
                             const float* __restrict__ Wdt,  // [128,4]
                             const float* __restrict__ bdt,  // [128]
                             float* __restrict__ dt) {
  int idx = blockIdx.x * blockDim.x + threadIdx.x;   // MTOT*128
  int m = idx >> 7, d = idx & 127;
  float v = bdt[d];
  #pragma unroll
  for (int r = 0; r < 4; ++r) v += dbl[m * DBLPAD + r] * Wdt[d * 4 + r];
  float sp = (v > 20.0f) ? v : log1pf(__expf(v));
  dt[m * 128 + d] = sp;
}

// ---------------------------------------------------------------------------
// (6) Selective scan with double-buffered async LDS staging.
// block = 512 = 32 channels x 16 states. Chunk = 32 timesteps:
//   sdt[32][32] @ 0, sx[32][32] @ 1024, sB[32][16] @ 2048, sC[32][16] @ 2560
//   (floats) = 12 KB/chunk, 2 buffers = 24 KB of the WGP's 320 KB LDS.
// Every wave issues exactly 2 async b128 copies per chunk (waves 8..15 mirror
// waves 0..7's B/C jobs) so a uniform s_wait_asynccnt(2) is correct.
// ---------------------------------------------------------------------------
__device__ inline void bm_scan_stage(const float* __restrict__ dt,
                                     const float* __restrict__ xcf,
                                     const float* __restrict__ dbl,
                                     float* buf, int m0, int dg, int tid) {
  {  // job A: dt rows (tid<256) / x rows (tid>=256), b128 per 4 channels
    int i = tid & 255;
    int t = i >> 3, q = i & 7;
    const float* src = (tid < 256) ? dt : xcf;
    const float* g = src + (size_t)(m0 + t) * 128 + dg * 32 + q * 4;
    float* l = buf + ((tid < 256) ? 0 : 1024) + t * 32 + q * 4;
    bm_cp128(g, l);
  }
  {  // job B: B (i<128) / C (i>=128); waves 8..15 duplicate waves 0..7
    int i = tid & 255;
    int t = (i & 127) >> 2, q = i & 3;
    int isC = (i >> 7) & 1;
    const float* g = dbl + (size_t)(m0 + t) * DBLPAD + (isC ? 20 : 4) + q * 4;
    float* l = buf + (isC ? 2560 : 2048) + t * 16 + q * 4;
    bm_cp128(g, l);
  }
}

__global__ void bm_scan_kernel(const float* __restrict__ xcf,
                               const float* __restrict__ dt,
                               const float* __restrict__ dbl,
                               const float* __restrict__ Aneg,  // [128,16]
                               const float* __restrict__ Dp,    // [128]
                               const float* __restrict__ xz,    // z at col 128
                               float* __restrict__ y, int rev) {
  __shared__ __align__(16) float sbuf[2][3072];
  int tid  = threadIdx.x;
  int nb   = blockIdx.x >> 2;
  int dg   = blockIdx.x & 3;
  int s    = tid & 15;
  int dloc = tid >> 4;             // 0..31
  int d    = dg * 32 + dloc;
  float A  = Aneg[d * 16 + s];
  float Dd = Dp[d];
  float h  = 0.0f;
  int mbase = nb * 512;

  bm_scan_stage(dt, xcf, dbl, &sbuf[0][0], mbase, dg, tid);  // prologue

  for (int ck = 0; ck < SCAN_NCH; ++ck) {
    if (ck + 1 < SCAN_NCH) {
      bm_scan_stage(dt, xcf, dbl, &sbuf[(ck + 1) & 1][0],
                    mbase + (ck + 1) * SCAN_T, dg, tid);
#if BM_ASYNC
      __builtin_amdgcn_s_wait_asynccnt(2);  // chunk ck complete, ck+1 in flight
#endif
    } else {
#if BM_ASYNC
      __builtin_amdgcn_s_wait_asynccnt(0);
#endif
    }
    __syncthreads();
    const float* cb = &sbuf[ck & 1][0];
    for (int tt = 0; tt < SCAN_T; ++tt) {
      float dtt = cb[tt * 32 + dloc];
      float xt  = cb[1024 + tt * 32 + dloc];
      float Bv  = cb[2048 + tt * 16 + s];
      float Cv  = cb[2560 + tt * 16 + s];
      float dA  = __expf(dtt * A);
      h = h * dA + (dtt * xt) * Bv;
      float c = h * Cv;
      c += __shfl_xor(c, 1, 16);
      c += __shfl_xor(c, 2, 16);
      c += __shfl_xor(c, 4, 16);
      c += __shfl_xor(c, 8, 16);
      if (s == 0) {
        int t = ck * SCAN_T + tt;
        int m = mbase + t;
        float zt = xz[m * 256 + 128 + d];
        float g  = zt * (1.0f / (1.0f + __expf(-zt)));
        int tn = rev ? (511 - t) : t;
        y[(mbase + tn) * 128 + d] = (c + Dd * xt) * g;
      }
    }
    __syncthreads();   // all waves done reading buf[ck&1] before it is refilled
  }
}

// ---------------------------------------------------------------------------
// (7) out_proj: out[MTOT,64] = (y_f + y_b)[MTOT,128] @ W_out^T, fused residual
// ---------------------------------------------------------------------------
__global__ void bm_outproj_kernel(const float* __restrict__ y0,
                                  const float* __restrict__ y1,
                                  const _Float16* __restrict__ Wo,  // [64,128] f16
                                  const float* __restrict__ xin,
                                  float* __restrict__ out) {
  int lane = threadIdx.x & 31;
  int wave = threadIdx.x >> 5;
  int tile = blockIdx.x * 8 + wave;        // 2048 * 4 tiles
  int mt = tile >> 2, nt = tile & 3;
  int kb = (lane < 16) ? 0 : 8;
  const float* a0 = y0 + (mt * 16 + (lane & 15)) * 128 + kb;
  const float* a1 = y1 + (mt * 16 + (lane & 15)) * 128 + kb;
  const _Float16* brow = Wo + (nt * 16 + (lane & 15)) * 128 + kb;
  v8f acc = {};
  #pragma unroll
  for (int kc = 0; kc < 4; ++kc) {
    v16h a;
    #pragma unroll
    for (int j = 0; j < 8; ++j) {
      a[j]     = (_Float16)(a0[kc * 32 + j]      + a1[kc * 32 + j]);
      a[8 + j] = (_Float16)(a0[kc * 32 + 16 + j] + a1[kc * 32 + 16 + j]);
    }
    v16h b = bm_frag_f16(brow + kc * 32);
    acc = __builtin_amdgcn_wmma_f32_16x16x32_f16(false, a, false, b,
                                                 (short)0, acc, false, false);
  }
  int c  = nt * 16 + (lane & 15);
  int mb = mt * 16 + ((lane < 16) ? 0 : 8);
  #pragma unroll
  for (int r = 0; r < 8; ++r) {
    int m = mb + r;
    int nb = m >> 9, t = m & 511;
    int pz = (nb >> 4) & 3, ph = (nb >> 2) & 3, pw = nb & 3;
    int nz = (t >> 6) & 7,  nh = (t >> 3) & 7,  nw = t & 7;
    int lin = ((c * 32 + nz * 4 + pz) * 32 + nh * 4 + ph) * 32 + nw * 4 + pw;
    out[lin] = acc[r] + xin[lin];
  }
}

// ---------------------------------------------------------------------------
extern "C" void kernel_launch(void* const* d_in, const int* in_sizes, int n_in,
                              void* d_out, int out_size, void* d_ws, size_t ws_size,
                              hipStream_t stream) {
  const float* x        = (const float*)d_in[0];
  const float* ln_gamma = (const float*)d_in[1];
  const float* ln_beta  = (const float*)d_in[2];
  const float* W_in[2]   = { (const float*)d_in[3],  (const float*)d_in[11] };
  const float* conv_w[2] = { (const float*)d_in[4],  (const float*)d_in[12] };
  const float* conv_b[2] = { (const float*)d_in[5],  (const float*)d_in[13] };
  const float* W_x[2]    = { (const float*)d_in[6],  (const float*)d_in[14] };
  const float* W_dt[2]   = { (const float*)d_in[7],  (const float*)d_in[15] };
  const float* b_dt[2]   = { (const float*)d_in[8],  (const float*)d_in[16] };
  const float* A_log[2]  = { (const float*)d_in[9],  (const float*)d_in[17] };
  const float* Dvec[2]   = { (const float*)d_in[10], (const float*)d_in[18] };
  const float* W_out     = (const float*)d_in[19];
  float* out = (float*)d_out;

  // workspace layout (256B aligned slices)
  char* ws = (char*)d_ws;
  size_t o = 0;
  auto take = [&](size_t bytes) -> char* {
    char* p = ws + o;
    o = (o + bytes + 255) & ~(size_t)255;
    return p;
  };
  _Float16* xnF16 = (_Float16*)take((size_t)MTOT * 64 * 2);
  _Float16* wInH  = (_Float16*)take((size_t)2 * 16384 * 2);
  _Float16* wXH   = (_Float16*)take((size_t)2 * 4608 * 2);
  _Float16* wOutH = (_Float16*)take((size_t)8192 * 2);
  float*    Aneg  = (float*)take((size_t)2 * 2048 * 4);
  float*    xz    = (float*)take((size_t)MTOT * 256 * 4);
  float*    xcf   = (float*)take((size_t)MTOT * 128 * 4);
  _Float16* xch   = (_Float16*)take((size_t)MTOT * 128 * 2);
  float*    dbl   = (float*)take((size_t)MTOT * DBLPAD * 4);
  float*    dtb   = (float*)take((size_t)MTOT * 128 * 4);
  float*    y0    = (float*)take((size_t)MTOT * 128 * 4);
  float*    y1    = (float*)take((size_t)MTOT * 128 * 4);
  (void)ws_size; (void)n_in; (void)in_sizes; (void)out_size;

  bm_prep_kernel<<<(54272 + 255) / 256, 256, 0, stream>>>(
      W_in[0], W_in[1], W_x[0], W_x[1], W_out, A_log[0], A_log[1],
      wInH, wXH, wOutH, Aneg);

  bm_ln_kernel<<<MTOT / 4, 256, 0, stream>>>(x, ln_gamma, ln_beta, xnF16);

  for (int dir = 0; dir < 2; ++dir) {
    bm_inproj_kernel<<<4096, 256, 0, stream>>>(xnF16, wInH + dir * 16384, xz, dir);
    bm_conv_kernel<<<(MTOT * 128) / 256, 256, 0, stream>>>(
        xz, conv_w[dir], conv_b[dir], xcf, xch);
    bm_xproj_kernel<<<768, 256, 0, stream>>>(xch, wXH + dir * 4608, dbl);
    bm_dt_kernel<<<(MTOT * 128) / 256, 256, 0, stream>>>(
        dbl, W_dt[dir], b_dt[dir], dtb);
    bm_scan_kernel<<<NBATCH * 4, 512, 0, stream>>>(
        xcf, dtb, dbl, Aneg + dir * 2048, Dvec[dir], xz,
        dir ? y1 : y0, dir);
  }

  bm_outproj_kernel<<<1024, 256, 0, stream>>>(y0, y1, wOutH, x, out);
}
